// GAT_48387101557055
// MI455X (gfx1250) — matrix operand order repaired and checked by reading, not compiled, not run on hardware
//
#include <hip/hip_runtime.h>
#include <hip/hip_bf16.h>

typedef _Float16 f16;
typedef __attribute__((ext_vector_type(8)))  _Float16 v8h;
typedef __attribute__((ext_vector_type(16))) _Float16 v16h;
typedef __attribute__((ext_vector_type(8)))  float    v8f;

#define NN        10000
#define NE        320000
#define IN_DIM    512
#define HID       64
#define H0        8
#define OUT_D     64
#define NEG_SLOPE 0.2f

// ---------------------------------------------------------------- utilities

__global__ void fill_u32_kernel(unsigned* __restrict__ p, unsigned v, long long n) {
  long long i = (long long)blockIdx.x * blockDim.x + threadIdx.x;
  if (i < n) p[i] = v;
}

__global__ void cvt_f16_kernel(const float* __restrict__ in, f16* __restrict__ out, long long n) {
  long long i = (long long)blockIdx.x * blockDim.x + threadIdx.x;
  if (i < n) out[i] = (f16)in[i];
}

__global__ void elu_cvt_f16_kernel(const float* __restrict__ in, f16* __restrict__ out, long long n) {
  long long i = (long long)blockIdx.x * blockDim.x + threadIdx.x;
  if (i < n) {
    float x = in[i];
    float y = x > 0.f ? x : (__expf(x) - 1.f);
    out[i] = (f16)y;
  }
}

// Pack f32 B[K,N] into per-wave WMMA fragment layout (f16):
// frag f = kt*(N/16)+nt holds 512 halves: [lane 0..31][slot 0..15]
//   k = kt*32 + (slot&7) + 8*(lane>=16) + 16*(slot>=8), n = nt*16 + (lane&15)
__global__ void pack_b_kernel(const float* __restrict__ B, f16* __restrict__ out,
                              int K, int N) {
  int i = blockIdx.x * blockDim.x + threadIdx.x;
  int nTiles = N >> 4;
  int total = (K >> 5) * nTiles * 512;
  if (i >= total) return;
  int s    = i & 15;
  int lane = (i >> 4) & 31;
  int f    = i >> 9;
  int nt   = f % nTiles;
  int kt   = f / nTiles;
  int k = (kt << 5) + (s & 7) + ((lane & 16) ? 8 : 0) + ((s & 8) ? 16 : 0);
  int n = (nt << 4) + (lane & 15);
  out[i] = (f16)B[(size_t)k * N + n];
}

__device__ __forceinline__ float leaky(float v) { return v > 0.f ? v : NEG_SLOPE * v; }

// monotonic float<->uint encoding so atomicMax(u32) implements float max
__device__ __forceinline__ unsigned encF(float f) {
  unsigned u = __float_as_uint(f);
  return (u & 0x80000000u) ? ~u : (u | 0x80000000u);
}
__device__ __forceinline__ float decF(unsigned u) {
  return (u & 0x80000000u) ? __uint_as_float(u & 0x7FFFFFFFu) : __uint_as_float(~u);
}

// ---------------------------------------------------------------- WMMA GEMM
// C[M,N] = A[M,K] x B[K,N]; A f16 row-major, Bp pre-packed fragments, C f32.
// M % 16 == 0, K % 32 == 0, N % 64 == 0. One wave -> 16x64 output strip.
__global__ __launch_bounds__(128)
void wmma_gemm_kernel(const f16* __restrict__ A, const f16* __restrict__ Bp,
                      float* __restrict__ C, int M, int K, int N) {
  int wave = blockIdx.x * 4 + (threadIdx.x >> 5);
  int lane = threadIdx.x & 31;
  int mTiles  = M >> 4;
  int nGroups = N >> 6;
  int nTiles  = N >> 4;
  if (wave >= mTiles * nGroups) return;
  int mt = wave / nGroups, ng = wave - mt * nGroups;
  int m0 = mt << 4;
  int n0 = ng << 6;
  int laneHi = (lane >> 4) & 1;   // 0: lanes 0-15, 1: lanes 16-31
  int laneLo = lane & 15;

  v8f acc[4] = {v8f{}, v8f{}, v8f{}, v8f{}};

  // A fragment: lane holds row m0+laneLo; halves k..k+7 and k+16..k+23 (+8 for hi lanes)
  const f16* arow  = A + (size_t)(m0 + laneLo) * K + laneHi * 8;
  const f16* bbase = Bp + ((size_t)ng * 4) * 512 + lane * 16;
  const int kSteps = K >> 5;

  for (int kt = 0; kt < kSteps; ++kt) {
    v8h alo = *(const v8h*)(arow);        // global_load_b128
    v8h ahi = *(const v8h*)(arow + 16);   // global_load_b128
    v16h a = __builtin_shufflevector(alo, ahi,
                                     0, 1, 2, 3, 4, 5, 6, 7,
                                     8, 9, 10, 11, 12, 13, 14, 15);
    const f16* bk = bbase + (size_t)kt * nTiles * 512;
#pragma unroll
    for (int t = 0; t < 4; ++t) {
      v16h b = *(const v16h*)(bk + (size_t)t * 512);   // 2x global_load_b128
      acc[t] = __builtin_amdgcn_wmma_f32_16x16x32_f16(
          false, a, false, b, (short)0, acc[t], false, false);
    }
    arow += 32;
  }

#pragma unroll
  for (int t = 0; t < 4; ++t) {
#pragma unroll
    for (int r = 0; r < 8; ++r) {
      int m = m0 + r + laneHi * 8;
      C[(size_t)m * N + n0 + t * 16 + laneLo] = acc[t][r];
    }
  }
}

// ---------------------------------------------------------------- GAT pieces

// el[n,h] = dot(feat[n,h,:], attn_l[h,:]);  er likewise
__global__ void elr_kernel(const float* __restrict__ feat, const float* __restrict__ al,
                           const float* __restrict__ ar, float* __restrict__ el,
                           float* __restrict__ er, int nNodes, int Hh, int D) {
  int i = blockIdx.x * blockDim.x + threadIdx.x;
  if (i >= nNodes * Hh) return;
  int h = i % Hh;
  const float* f = feat + (size_t)i * D;
  const float* l = al + (size_t)h * D;
  const float* r = ar + (size_t)h * D;
  float sl = 0.f, sr = 0.f;
  for (int d = 0; d < D; ++d) { float v = f[d]; sl += v * l[d]; sr += v * r[d]; }
  el[i] = sl;
  er[i] = sr;
}

// pass 1: per-dst running max of leaky(el[src]+er[dst])
__global__ void edge_max_kernel(const int* __restrict__ src, const int* __restrict__ dst,
                                const float* __restrict__ el, const float* __restrict__ er,
                                unsigned* __restrict__ mEnc, int E, int Hh) {
  int i = blockIdx.x * blockDim.x + threadIdx.x;
  if (i >= E * Hh) return;
  int e = i / Hh, h = i - e * Hh;
  float v = leaky(el[(size_t)src[e] * Hh + h] + er[(size_t)dst[e] * Hh + h]);
  atomicMax(mEnc + (size_t)dst[e] * Hh + h, encF(v));
}

// pass 2: ex = exp(e - m[dst]); s[dst] += ex
__global__ void edge_expsum_kernel(const int* __restrict__ src, const int* __restrict__ dst,
                                   const float* __restrict__ el, const float* __restrict__ er,
                                   const unsigned* __restrict__ mEnc, float* __restrict__ ex,
                                   float* __restrict__ sSum, int E, int Hh) {
  int i = blockIdx.x * blockDim.x + threadIdx.x;
  if (i >= E * Hh) return;
  int e = i / Hh, h = i - e * Hh;
  int d = dst[e];
  float v = leaky(el[(size_t)src[e] * Hh + h] + er[(size_t)d * Hh + h]);
  float x = __expf(v - decF(mEnc[(size_t)d * Hh + h]));
  ex[i] = x;
  unsafeAtomicAdd(sSum + (size_t)d * Hh + h, x);
}

// pass 3: out[dst,h,:] += (ex/s[dst]) * feat[src,h,:]   (one thread per 4 dims)
__global__ void edge_agg_kernel(const int* __restrict__ src, const int* __restrict__ dst,
                                const float* __restrict__ ex, const float* __restrict__ sSum,
                                const float* __restrict__ feat, float* __restrict__ out,
                                int E, int Hh, int D) {
  long long i = (long long)blockIdx.x * blockDim.x + threadIdx.x;
  int quads = D >> 2;
  long long total = (long long)E * Hh * quads;
  if (i >= total) return;
  int q = (int)(i % quads);
  long long eh = i / quads;
  int h = (int)(eh % Hh);
  int e = (int)(eh / Hh);
  float alpha = ex[eh] / sSum[(size_t)dst[e] * Hh + h];
  const float4 v = ((const float4*)(feat + ((size_t)src[e] * Hh + h) * D))[q];
  float* o = out + ((size_t)dst[e] * Hh + h) * D + (size_t)q * 4;
  unsafeAtomicAdd(o + 0, alpha * v.x);
  unsafeAtomicAdd(o + 1, alpha * v.y);
  unsafeAtomicAdd(o + 2, alpha * v.z);
  unsafeAtomicAdd(o + 3, alpha * v.w);
}

// ---------------------------------------------------------------- launch

extern "C" void kernel_launch(void* const* d_in, const int* in_sizes, int n_in,
                              void* d_out, int out_size, void* d_ws, size_t ws_size,
                              hipStream_t stream) {
  const float* h   = (const float*)d_in[0];
  const int*   src = (const int*)d_in[1];
  const int*   dst = (const int*)d_in[2];
  const float* W1  = (const float*)d_in[3];
  const float* al1 = (const float*)d_in[4];
  const float* ar1 = (const float*)d_in[5];
  const float* W2  = (const float*)d_in[6];
  const float* al2 = (const float*)d_in[7];
  const float* ar2 = (const float*)d_in[8];
  float* out2 = (float*)d_out;            // (NN, 64) fp32

  char* ws = (char*)d_ws;
  size_t off = 0;
  auto alloc = [&](size_t bytes) -> void* {
    off = (off + 255) & ~(size_t)255;
    void* p = ws + off;
    off += bytes;
    return p;
  };

  f16*      A1    = (f16*)alloc((size_t)NN * IN_DIM * sizeof(f16));
  f16*      W1p   = (f16*)alloc((size_t)IN_DIM * (H0 * HID) * sizeof(f16));   // packed frags
  float*    feat1 = (float*)alloc((size_t)NN * H0 * HID * sizeof(float));
  float*    el1   = (float*)alloc((size_t)NN * H0 * sizeof(float));
  float*    er1   = (float*)alloc((size_t)NN * H0 * sizeof(float));
  unsigned* m1    = (unsigned*)alloc((size_t)NN * H0 * sizeof(unsigned));
  float*    s1    = (float*)alloc((size_t)NN * H0 * sizeof(float));
  float*    ex1   = (float*)alloc((size_t)NE * H0 * sizeof(float));
  float*    out1  = (float*)alloc((size_t)NN * H0 * HID * sizeof(float));
  f16*      X16   = (f16*)alloc((size_t)NN * H0 * HID * sizeof(f16));
  f16*      W2p   = (f16*)alloc((size_t)(H0 * HID) * OUT_D * sizeof(f16));    // packed frags
  float*    feat2 = (float*)alloc((size_t)NN * OUT_D * sizeof(float));
  float*    el2   = (float*)alloc((size_t)NN * sizeof(float));
  float*    er2   = (float*)alloc((size_t)NN * sizeof(float));
  unsigned* m2    = (unsigned*)alloc((size_t)NN * sizeof(unsigned));
  float*    s2    = (float*)alloc((size_t)NN * sizeof(float));
  float*    ex2   = (float*)alloc((size_t)NE * sizeof(float));

  const int T = 256;
  auto g = [](long long n, int t) { return dim3((unsigned)((n + t - 1) / t)); };

  // convert A to f16; pack B matrices into WMMA fragment layout
  cvt_f16_kernel<<<g((long long)NN * IN_DIM, T), T, 0, stream>>>(h, A1, (long long)NN * IN_DIM);
  pack_b_kernel<<<g((long long)IN_DIM * H0 * HID, T), T, 0, stream>>>(W1, W1p, IN_DIM, H0 * HID);
  pack_b_kernel<<<g((long long)H0 * HID * OUT_D, T), T, 0, stream>>>(W2, W2p, H0 * HID, OUT_D);

  // init accumulators: m encodings (0 == smaller than any encoded float), sums, outputs
  fill_u32_kernel<<<g(NN * H0, T), T, 0, stream>>>(m1, 0u, NN * H0);
  fill_u32_kernel<<<g(NN * H0, T), T, 0, stream>>>((unsigned*)s1, 0u, NN * H0);
  fill_u32_kernel<<<g((long long)NN * H0 * HID, T), T, 0, stream>>>((unsigned*)out1, 0u, (long long)NN * H0 * HID);
  fill_u32_kernel<<<g(NN, T), T, 0, stream>>>(m2, 0u, NN);
  fill_u32_kernel<<<g(NN, T), T, 0, stream>>>((unsigned*)s2, 0u, NN);
  fill_u32_kernel<<<g((long long)NN * OUT_D, T), T, 0, stream>>>((unsigned*)out2, 0u, (long long)NN * OUT_D);

  // ---- layer 1 ----
  {
    int waves = (NN / 16) * ((H0 * HID) / 64);   // 625 * 8 = 5000
    wmma_gemm_kernel<<<dim3((waves + 3) / 4), dim3(128), 0, stream>>>(
        A1, W1p, feat1, NN, IN_DIM, H0 * HID);
  }
  elr_kernel<<<g(NN * H0, T), T, 0, stream>>>(feat1, al1, ar1, el1, er1, NN, H0, HID);
  edge_max_kernel<<<g((long long)NE * H0, T), T, 0, stream>>>(src, dst, el1, er1, m1, NE, H0);
  edge_expsum_kernel<<<g((long long)NE * H0, T), T, 0, stream>>>(src, dst, el1, er1, m1, ex1, s1, NE, H0);
  edge_agg_kernel<<<g((long long)NE * H0 * (HID / 4), T), T, 0, stream>>>(
      src, dst, ex1, s1, feat1, out1, NE, H0, HID);

  // ELU activation + convert to f16 as layer-2 GEMM input
  elu_cvt_f16_kernel<<<g((long long)NN * H0 * HID, T), T, 0, stream>>>(out1, X16, (long long)NN * H0 * HID);

  // ---- layer 2 ----
  {
    int waves = (NN / 16) * (OUT_D / 64);        // 625
    wmma_gemm_kernel<<<dim3((waves + 3) / 4), dim3(128), 0, stream>>>(
        X16, W2p, feat2, NN, H0 * HID, OUT_D);
  }
  elr_kernel<<<g(NN, T), T, 0, stream>>>(feat2, al2, ar2, el2, er2, NN, 1, OUT_D);
  edge_max_kernel<<<g(NE, T), T, 0, stream>>>(src, dst, el2, er2, m2, NE, 1);
  edge_expsum_kernel<<<g(NE, T), T, 0, stream>>>(src, dst, el2, er2, m2, ex2, s2, NE, 1);
  // H1 == 1, so mean over heads is identity: aggregate straight into d_out
  edge_agg_kernel<<<g((long long)NE * (OUT_D / 4), T), T, 0, stream>>>(
      src, dst, ex2, s2, feat2, out2, NE, 1, OUT_D);
}
